// Net_326417514748
// MI455X (gfx1250) — matrix operand order repaired and checked by reading, not compiled, not run on hardware
//
#include <hip/hip_runtime.h>
#include <hip/hip_bf16.h>

// ---------------- problem constants (fixed by reference) ----------------
#define NN 50000
#define EE 320000
#define HH 256
#define LL 5
#define GG 512
#define SS 5
#define VV 5002
#define VP 5008          // head_W bf16 padded leading dim (16B-aligned rows)
#define DD (LL*HH)       // 1280

// ---------------- WMMA types ----------------
typedef __attribute__((ext_vector_type(16))) __bf16 v16bf;
typedef __attribute__((ext_vector_type(8)))  float  v8f;

// ---------------- CDNA5 async global->LDS copy (16B per lane) ----------------
// Builtin prototype (from compiler diagnostic): first param is
// 'int __attribute__((vector_size(16))) __device__ *' (AS1, non-const),
// second is the AS3 (LDS) pointer to the same vector type, then imm offset, imm cpol.
typedef int v4i_ __attribute__((vector_size(16)));

#if __has_builtin(__builtin_amdgcn_global_load_async_to_lds_b128)
#define ASYNC_COPY16(gptr, lptr)                                                   \
    __builtin_amdgcn_global_load_async_to_lds_b128(                                \
        (__attribute__((address_space(1))) v4i_*)(gptr),                           \
        (__attribute__((address_space(3))) v4i_*)(lptr), 0, 0)
#else
#define ASYNC_COPY16(gptr, lptr)                                                   \
    asm volatile("global_load_async_to_lds_b128 %0, %1, off"                       \
                 :: "v"((unsigned)(unsigned long long)(lptr)),                     \
                    "v"((unsigned long long)(gptr)) : "memory")
#endif

#if __has_builtin(__builtin_amdgcn_s_wait_asynccnt)
#define WAIT_ASYNC() __builtin_amdgcn_s_wait_asynccnt(0)
#else
#define WAIT_ASYNC() asm volatile("s_wait_asynccnt 0" ::: "memory")
#endif

__device__ __forceinline__ unsigned short f2bf(float f) {
    unsigned int u = __float_as_uint(f);
    u = (u + 0x7FFFu + ((u >> 16) & 1u)) >> 16;   // round-to-nearest-even
    return (unsigned short)u;
}

// ---------------- elementwise kernels ----------------
__global__ void k_zero(float* p, long long n) {
    long long i = (long long)blockIdx.x * blockDim.x + threadIdx.x;
    if (i < n) p[i] = 0.f;
}

__global__ void k_cvt_bf16(const float* __restrict__ in, unsigned short* __restrict__ out, long long n) {
    long long i = (long long)blockIdx.x * blockDim.x + threadIdx.x;
    if (i < n) out[i] = f2bf(in[i]);
}

// head_W [S][D][V] fp32 -> [S][D][VP] bf16 (zero padded)
__global__ void k_cvt_pad(const float* __restrict__ in, unsigned short* __restrict__ out) {
    long long i = (long long)blockIdx.x * blockDim.x + threadIdx.x;
    if (i >= (long long)SS * DD * VP) return;
    int v = (int)(i % VP);
    long long sd = i / VP;
    out[i] = (v < VV) ? f2bf(in[sd * VV + v]) : (unsigned short)0;
}

// node encoder: h0 = emb_node0[x0] + emb_node1[x1] + emb_depth[d]
__global__ void k_encode(const int* __restrict__ x, const int* __restrict__ dep,
                         const float* __restrict__ e0, const float* __restrict__ e1,
                         const float* __restrict__ ed, float* __restrict__ h0) {
    long long i = (long long)blockIdx.x * blockDim.x + threadIdx.x;
    if (i >= (long long)NN * HH) return;
    long long n = i >> 8;  int j = (int)(i & 255);
    h0[i] = e0[(long long)x[2*n] * HH + j]
          + e1[(long long)x[2*n+1] * HH + j]
          + ed[(long long)dep[n] * HH + j];
}

// z = (1+eps)*h + agg, cast bf16
__global__ void k_combine(const float* __restrict__ hin, int hstride,
                          const float* __restrict__ agg,
                          const float* __restrict__ eps, int l,
                          unsigned short* __restrict__ z) {
    long long i = (long long)blockIdx.x * blockDim.x + threadIdx.x;
    if (i >= (long long)NN * HH) return;
    long long n = i >> 8;  int j = (int)(i & 255);
    float v = (1.f + eps[l]) * hin[n * (long long)hstride + j] + agg[i];
    z[i] = f2bf(v);
}

// ---------------- edge message + scatter-add ----------------
// e = edge_attr @ eW + eb (eW 8x256 cached in LDS); msg = relu(h[src]+e); agg[dst] += msg
__global__ __launch_bounds__(256) void k_edge(
    const int* __restrict__ ei, const float* __restrict__ ea,
    const float* __restrict__ eW, const float* __restrict__ eb,
    const float* __restrict__ hin, int hstride, float* __restrict__ agg)
{
    __shared__ __align__(16) float sW[8 * HH];
    __shared__ __align__(16) float sb[HH];
    for (int i = threadIdx.x; i < 8 * HH; i += 256) sW[i] = eW[i];
    for (int i = threadIdx.x; i < HH;     i += 256) sb[i] = eb[i];
    __syncthreads();

    const int le = threadIdx.x >> 6;        // 4 edges in flight per block
    const int j  = (threadIdx.x & 63) << 2; // 4 features per thread
    const long long estep = (long long)gridDim.x * 4;
    for (long long e = (long long)blockIdx.x * 4 + le; e < EE; e += estep) {
        if (e + estep < EE) __builtin_prefetch(ea + (e + estep) * 8, 0, 1);
        const int s = ei[e];
        const int d = ei[EE + e];
        float4 a0 = *(const float4*)(ea + e * 8);
        float4 a1 = *(const float4*)(ea + e * 8 + 4);
        float ak[8] = {a0.x, a0.y, a0.z, a0.w, a1.x, a1.y, a1.z, a1.w};
        float4 v = *(const float4*)&sb[j];
        #pragma unroll
        for (int k = 0; k < 8; k++) {
            float4 w = *(const float4*)&sW[k * HH + j];
            v.x += ak[k] * w.x; v.y += ak[k] * w.y;
            v.z += ak[k] * w.z; v.w += ak[k] * w.w;
        }
        float4 h = *(const float4*)(hin + (long long)s * hstride + j);
        float m0 = fmaxf(h.x + v.x, 0.f);
        float m1 = fmaxf(h.y + v.y, 0.f);
        float m2 = fmaxf(h.z + v.z, 0.f);
        float m3 = fmaxf(h.w + v.w, 0.f);
        float* ap = agg + (long long)d * HH + j;
        atomicAdd(ap + 0, m0);
        atomicAdd(ap + 1, m1);
        atomicAdd(ap + 2, m2);
        atomicAdd(ap + 3, m3);
    }
}

// ---------------- deterministic graph mean-pool (batch is sorted) ----------------
__global__ __launch_bounds__(256) void k_pool(const float* __restrict__ xs,
                                              const int* __restrict__ batch,
                                              float* __restrict__ pooled) {
    int g = blockIdx.y;
    int lo = 0, hi = NN;
    while (lo < hi) { int mid = (lo + hi) >> 1; if (batch[mid] < g) lo = mid + 1; else hi = mid; }
    int start = lo;
    hi = NN;
    while (lo < hi) { int mid = (lo + hi) >> 1; if (batch[mid] < g + 1) lo = mid + 1; else hi = mid; }
    int end = lo;
    int col = blockIdx.x * 256 + threadIdx.x;
    if (col >= DD) return;
    float sum = 0.f;
    for (int n = start; n < end; n++) sum += xs[(long long)n * DD + col];
    float c = (float)(end - start);
    pooled[(long long)g * DD + col] = sum / fmaxf(c, 1.f);
}

// ---------------- bf16 WMMA GEMM ----------------
// out[z][m][n] = act(A[m][:] . B[z][:][n] + bias[z][n]); A shared across z.
// Block tile 128x128, 8 waves (wave32), wave tile 64x32 = 4x2 WMMA 16x16 tiles, K-step 32.
// A tile staged with GLOBAL_LOAD_ASYNC_TO_LDS_B128 when fully in-range (no zero-fill needed).
#define BM 128
#define BN 128
#define TK 32
#define APAD 40   // ushorts per LDS row (32 + 8), keeps 16B alignment

__global__ __launch_bounds__(256) void wmma_gemm(
    const unsigned short* __restrict__ A, int lda,
    const unsigned short* __restrict__ B, int ldb, long long strideB,
    const float* __restrict__ bias, long long strideBias,
    float* __restrict__ outF, unsigned short* __restrict__ outBf,
    int ldo, long long strideOut,
    int M, int Nn, int Kd, int doRelu)
{
    __shared__ __align__(16) unsigned short Asm[BM * APAD];
    __shared__ __align__(16) unsigned short Bsm[BN * APAD];

    const int tid    = threadIdx.x;
    const int lane   = tid & 31;
    const int wave   = tid >> 5;
    const int wr     = wave >> 2;          // 0..1
    const int wc     = wave & 3;           // 0..3
    const int half16 = (lane >> 4) & 1;
    const int lmod   = lane & 15;
    const int mBase  = blockIdx.y * BM;
    const int nBase  = blockIdx.x * BN;
    const unsigned short* Bz = B + (long long)blockIdx.z * strideB;
    const bool fullA = (mBase + BM) <= M;   // block-uniform

    v8f acc[4][2] = {};

    for (int k0 = 0; k0 < Kd; k0 += TK) {
        // ---- A tile: BM x TK row-major ----
        if (fullA) {
            // CDNA5 async copy straight into LDS (ASYNCcnt-tracked)
            for (int u = tid; u < (BM * TK) / 8; u += 256) {
                int row = u >> 2;
                int c8  = (u & 3) << 3;
                ASYNC_COPY16(A + (long long)(mBase + row) * lda + k0 + c8,
                             &Asm[row * APAD + c8]);
            }
        } else {
            for (int u = tid; u < (BM * TK) / 8; u += 256) {
                int row = u >> 2;
                int c8  = (u & 3) << 3;
                int gr  = mBase + row;
                uint4 val = {0u, 0u, 0u, 0u};
                if (gr < M) val = *(const uint4*)(A + (long long)gr * lda + k0 + c8);
                *(uint4*)(&Asm[row * APAD + c8]) = val;
            }
        }
        // ---- B tile: TK x BN row-major in memory -> transposed [n][k] in LDS ----
        for (int u = tid; u < (TK * BN) / 8; u += 256) {
            int kk = u >> 4;
            int n8 = (u & 15) << 3;
            int gn = nBase + n8;
            unsigned short tmp[8] __attribute__((aligned(16)));
            if (gn + 8 <= Nn) {
                *(uint4*)tmp = *(const uint4*)(Bz + (long long)(k0 + kk) * ldb + gn);
            } else {
                #pragma unroll
                for (int t = 0; t < 8; t++)
                    tmp[t] = (gn + t < Nn) ? Bz[(long long)(k0 + kk) * ldb + gn + t] : (unsigned short)0;
            }
            #pragma unroll
            for (int t = 0; t < 8; t++) Bsm[(n8 + t) * APAD + kk] = tmp[t];
        }
        if (fullA) WAIT_ASYNC();
        __syncthreads();

        // A fragment: lane<16: K=0..7 (v0-3), 16..23 (v4-7); lane>=16: +8
        v16bf af[4], bfv[2];
        #pragma unroll
        for (int i = 0; i < 4; i++) {
            const unsigned short* p = &Asm[(wr * 64 + i * 16 + lmod) * APAD];
            union { v16bf v; uint4 q[2]; } u;
            u.q[0] = *(const uint4*)(p + 8 * half16);
            u.q[1] = *(const uint4*)(p + 16 + 8 * half16);
            af[i] = u.v;
        }
        // B fragment: N = lane%16, K = same as A (0..7,16..23; lane>=16: +8); transposed LDS [n][k]
        #pragma unroll
        for (int jj = 0; jj < 2; jj++) {
            const unsigned short* p = &Bsm[(wc * 32 + jj * 16 + lmod) * APAD];
            union { v16bf v; uint4 q[2]; } u;
            u.q[0] = *(const uint4*)(p + 8 * half16);
            u.q[1] = *(const uint4*)(p + 16 + 8 * half16);
            bfv[jj] = u.v;
        }
        #pragma unroll
        for (int i = 0; i < 4; i++)
            #pragma unroll
            for (int jj = 0; jj < 2; jj++)
                acc[i][jj] = __builtin_amdgcn_wmma_f32_16x16x32_bf16(
                    false, af[i], false, bfv[jj], (short)0, acc[i][jj], false, false);
        __syncthreads();
    }

    // Epilogue: C/D layout row = r + 8*(lane>=16), col = lane%16
    const float* biz = bias + (long long)blockIdx.z * strideBias;
    #pragma unroll
    for (int i = 0; i < 4; i++) {
        #pragma unroll
        for (int jj = 0; jj < 2; jj++) {
            union { v8f v; float f[8]; } u; u.v = acc[i][jj];
            #pragma unroll
            for (int r = 0; r < 8; r++) {
                int gr = mBase + wr * 64 + i * 16 + half16 * 8 + r;
                int gc = nBase + wc * 32 + jj * 16 + lmod;
                if (gr < M && gc < Nn) {
                    float v = u.f[r] + biz[gc];
                    if (doRelu) v = fmaxf(v, 0.f);
                    long long idx = (long long)blockIdx.z * strideOut + (long long)gr * ldo + gc;
                    if (outBf) outBf[idx] = f2bf(v);
                    else       outF[idx]  = v;
                }
            }
        }
    }
}

// ---------------- host-side orchestration ----------------
extern "C" void kernel_launch(void* const* d_in, const int* in_sizes, int n_in,
                              void* d_out, int out_size, void* d_ws, size_t ws_size,
                              hipStream_t stream) {
    (void)in_sizes; (void)n_in; (void)out_size; (void)ws_size;
    const int*   x      = (const int*)d_in[0];
    const int*   depth  = (const int*)d_in[1];
    const int*   ei     = (const int*)d_in[2];
    const float* ea     = (const float*)d_in[3];
    const int*   batch  = (const int*)d_in[4];
    const float* e0     = (const float*)d_in[5];
    const float* e1     = (const float*)d_in[6];
    const float* ed     = (const float*)d_in[7];
    const float* edgeW  = (const float*)d_in[8];
    const float* edgeB  = (const float*)d_in[9];
    const float* eps    = (const float*)d_in[10];
    const float* W1     = (const float*)d_in[11];
    const float* b1     = (const float*)d_in[12];
    const float* W2     = (const float*)d_in[13];
    const float* b2     = (const float*)d_in[14];
    const float* headW  = (const float*)d_in[15];
    const float* headB  = (const float*)d_in[16];
    float*       out    = (float*)d_out;

    char* ws = (char*)d_ws;
    size_t off = 0;
    auto carve = [&](size_t bytes) { void* p = ws + off; off = (off + bytes + 255) & ~(size_t)255; return p; };
    float*          h0     = (float*)carve((size_t)NN * HH * 4);
    float*          agg    = (float*)carve((size_t)NN * HH * 4);
    unsigned short* zbf    = (unsigned short*)carve((size_t)NN * HH * 2);
    unsigned short* t1bf   = (unsigned short*)carve((size_t)NN * 2 * HH * 2);
    float*          xs     = (float*)carve((size_t)NN * DD * 4);
    float*          pooled = (float*)carve((size_t)GG * DD * 4);
    unsigned short* hgbf   = (unsigned short*)carve((size_t)GG * DD * 2);
    unsigned short* w1bf   = (unsigned short*)carve((size_t)LL * HH * 2 * HH * 2);
    unsigned short* w2bf   = (unsigned short*)carve((size_t)LL * 2 * HH * HH * 2);
    unsigned short* hwbf   = (unsigned short*)carve((size_t)SS * DD * VP * 2);

    const int T = 256;
    // weight conversions (deterministic, every call)
    long long nW = (long long)LL * HH * 2 * HH;
    k_cvt_bf16<<<(int)((nW + T - 1) / T), T, 0, stream>>>(W1, w1bf, nW);
    k_cvt_bf16<<<(int)((nW + T - 1) / T), T, 0, stream>>>(W2, w2bf, nW);
    long long nHW = (long long)SS * DD * VP;
    k_cvt_pad<<<(int)((nHW + T - 1) / T), T, 0, stream>>>(headW, hwbf);

    long long nNH = (long long)NN * HH;
    k_encode<<<(int)((nNH + T - 1) / T), T, 0, stream>>>(x, depth, e0, e1, ed, h0);

    for (int l = 0; l < LL; l++) {
        const float* hin = (l == 0) ? h0 : (xs + (l - 1) * HH);
        int hstride = (l == 0) ? HH : DD;

        k_zero<<<(int)((nNH + T - 1) / T), T, 0, stream>>>(agg, nNH);
        k_edge<<<2048, T, 0, stream>>>(ei, ea, edgeW + (long long)l * 8 * HH,
                                       edgeB + (long long)l * HH, hin, hstride, agg);
        k_combine<<<(int)((nNH + T - 1) / T), T, 0, stream>>>(hin, hstride, agg, eps, l, zbf);

        // GEMM1: [N,256] x [256,512] -> relu -> bf16
        dim3 g1((2 * HH + BN - 1) / BN, (NN + BM - 1) / BM, 1);
        wmma_gemm<<<g1, T, 0, stream>>>(zbf, HH,
                                        w1bf + (long long)l * HH * 2 * HH, 2 * HH, 0,
                                        b1 + (long long)l * 2 * HH, 0,
                                        nullptr, t1bf, 2 * HH, 0,
                                        NN, 2 * HH, HH, 1);
        // GEMM2: [N,512] x [512,256] -> fp32 into JK-cat slice (ldo = 1280)
        dim3 g2((HH + BN - 1) / BN, (NN + BM - 1) / BM, 1);
        wmma_gemm<<<g2, T, 0, stream>>>(t1bf, 2 * HH,
                                        w2bf + (long long)l * 2 * HH * HH, HH, 0,
                                        b2 + (long long)l * HH, 0,
                                        xs + (long long)l * HH, nullptr, DD, 0,
                                        NN, HH, 2 * HH, 0);
    }

    // mean pool (deterministic: binary search on sorted batch)
    k_pool<<<dim3((DD + T - 1) / T, GG), T, 0, stream>>>(xs, batch, pooled);
    long long nGD = (long long)GG * DD;
    k_cvt_bf16<<<(int)((nGD + T - 1) / T), T, 0, stream>>>(pooled, hgbf, nGD);

    // heads: batched GEMM [512,1280] x [1280,5002] over S
    dim3 gh((VV + BN - 1) / BN, (GG + BM - 1) / BM, SS);
    wmma_gemm<<<gh, T, 0, stream>>>(hgbf, DD,
                                    hwbf, VP, (long long)DD * VP,
                                    headB, VV,
                                    out, nullptr, VV, (long long)GG * VV,
                                    GG, VV, DD, 0);
}
